// AttnNN_85272280695389
// MI455X (gfx1250) — matrix operand order repaired and checked by reading, not compiled, not run on hardware
//
#include <hip/hip_runtime.h>
#include <hip/hip_bf16.h>

// ---------------- constants ----------------
#define D_EMB   128
#define HID     64
#define SEQL    21
#define BATCHN  64
#define NNODES  16384
#define NEDGES  262144
#define EHALF   131072
#define SEQROWS (BATCHN * SEQL)          // 1344
#define SEQSZ   (SEQROWS * D_EMB)        // 172032
#define NODESZ  (NNODES * D_EMB)         // 2097152
#define GATESZ  (SEQROWS * 4 * HID)      // 344064
#define HBUFSZ  (SEQROWS * HID)          // 86016
#define LN_EPS  1e-5f

// ---------------- WMMA types ----------------
typedef __attribute__((ext_vector_type(16))) unsigned short v16u;
typedef __attribute__((ext_vector_type(16))) __bf16         v16bf;
typedef __attribute__((ext_vector_type(8)))  float          v8f;

union frag32 { v16u v; uint4 q[2]; };

__device__ __forceinline__ unsigned short f2bfu(float f) {
    unsigned int u = __builtin_bit_cast(unsigned int, f);
    unsigned int r = u + 0x7FFFu + ((u >> 16) & 1u);   // round-to-nearest-even
    return (unsigned short)(r >> 16);
}
__device__ __forceinline__ float bfu2f(unsigned short h) {
    unsigned int u = ((unsigned int)h) << 16;
    return __builtin_bit_cast(float, u);
}
__device__ __forceinline__ v8f wmma_bf16(v16u a, v16u b, v8f c) {
    return __builtin_amdgcn_wmma_f32_16x16x32_bf16(
        false, __builtin_bit_cast(v16bf, a),
        false, __builtin_bit_cast(v16bf, b),
        (short)0, c, false, false);
}
__device__ __forceinline__ float sigm(float x) { return 1.f / (1.f + __expf(-x)); }

// ---------------- generic bf16-WMMA GEMM ----------------
// A: bf16, row-major, lda multiple of 8, zero-padded to Kpad (multiple of 32).
// Bfrag: fragment-order bf16: [kt][nt(NT total)][lane 0..31][16 elems].
// One wave computes a 32-row x 128-col tile (2 m-tiles x 8 n-tiles).
__global__ __launch_bounds__(32) void k_gemm(
    const unsigned short* __restrict__ A, int lda, int Kpad,
    const unsigned short* __restrict__ Bfrag, int NT,
    const float* __restrict__ bias,
    float* __restrict__ C, int ldc)
{
    const int row0 = blockIdx.x * 32;
    const int nt0  = blockIdx.y * 8;
    const int lane = threadIdx.x;
    const int mloc = lane & 15;
    const int hi   = lane >> 4;
    const unsigned short* arow0 = A + (size_t)(row0 + mloc) * lda;
    const unsigned short* arow1 = arow0 + (size_t)16 * lda;

    v8f acc[2][8];
#pragma unroll
    for (int i = 0; i < 2; ++i)
#pragma unroll
        for (int j = 0; j < 8; ++j)
#pragma unroll
            for (int l = 0; l < 8; ++l) acc[i][j][l] = 0.f;

    for (int kb = 0; kb < Kpad; kb += 32) {
        const int kt = kb >> 5;
        frag32 a0, a1;                               // 4x b128
        a0.q[0] = *(const uint4*)(arow0 + kb + hi * 8);
        a0.q[1] = *(const uint4*)(arow0 + kb + 16 + hi * 8);
        a1.q[0] = *(const uint4*)(arow1 + kb + hi * 8);
        a1.q[1] = *(const uint4*)(arow1 + kb + 16 + hi * 8);
        frag32 b[8];                                 // 16x b128, loaded up front
        const unsigned short* bp =
            Bfrag + (((size_t)(kt * NT + nt0) * 32 + lane) << 4);
#pragma unroll
        for (int nt = 0; nt < 8; ++nt) {
            b[nt].q[0] = *(const uint4*)(bp);
            b[nt].q[1] = *(const uint4*)(bp + 8);
            bp += 512;
        }
#pragma unroll
        for (int nt = 0; nt < 8; ++nt) {
            acc[0][nt] = wmma_bf16(a0.v, b[nt].v, acc[0][nt]);
            acc[1][nt] = wmma_bf16(a1.v, b[nt].v, acc[1][nt]);
        }
    }
#pragma unroll
    for (int mt = 0; mt < 2; ++mt)
#pragma unroll
        for (int nt = 0; nt < 8; ++nt) {
            int ch = (nt0 + nt) * 16 + mloc;
            float bv = bias[ch];
#pragma unroll
            for (int r = 0; r < 8; ++r) {
                int m = row0 + mt * 16 + r + hi * 8;
                C[(size_t)m * ldc + ch] = acc[mt][nt][r] + bv;
            }
        }
}

// ---------------- fused edge kernel ----------------
// For 32 edges: gather fs=nodes[src], fd=nodes[dst] (bf16) into LDS (b128),
// gate = sigmoid([fs|fd|ea] @ W.T + b) via WMMA (K 259 padded to 288),
// atomicAdd(gate*fd -> acc[src]) then (gate*fs -> acc[dst]).
__global__ __launch_bounds__(32) void k_edge(
    const unsigned short* __restrict__ nodes,     // N x 128 bf16
    const int* __restrict__ edge_index,           // 2 x NEDGES
    const float* __restrict__ edge_attr,          // NEDGES x 3
    const unsigned short* __restrict__ Bin,       // frag-order, KT=9, NT=8
    const unsigned short* __restrict__ Bout,
    const float* __restrict__ bin,
    const float* __restrict__ bout,
    float* __restrict__ acc)
{
    const int e0   = blockIdx.x * 32;
    const int lane = threadIdx.x;
    __shared__ __align__(16) unsigned short feat[32][320]; // fs|fd|ea_in(+pad)|ea_out(+pad)
    __shared__ int sidx[32], didx[32];

    {
        int e = e0 + lane;
        sidx[lane] = edge_index[2 * e];
        didx[lane] = edge_index[NEDGES + 2 * e];
        for (int c = 0; c < 64; ++c) feat[lane][256 + c] = 0;
        for (int c = 0; c < 3; ++c) {
            feat[lane][256 + c] = f2bfu(edge_attr[(2 * e + 1) * 3 + c]); // ea_in  = attr[1::2]
            feat[lane][288 + c] = f2bfu(edge_attr[(2 * e) * 3 + c]);     // ea_out = attr[0::2]
        }
    }
    __syncthreads();
    // gather: 32 edges x 128 cols, 8 bf16 (16B) per transfer
    for (int idx = lane; idx < 32 * 16; idx += 32) {
        int m = idx >> 4, c8 = (idx & 15) * 8;
        *(uint4*)&feat[m][c8]       = *(const uint4*)(nodes + (size_t)sidx[m] * 128 + c8);
        *(uint4*)&feat[m][128 + c8] = *(const uint4*)(nodes + (size_t)didx[m] * 128 + c8);
    }
    __syncthreads();

    const int mloc = lane & 15, hi = lane >> 4;

    for (int phase = 0; phase < 2; ++phase) {
        const unsigned short* B = phase ? Bout : Bin;
        const float* bias       = phase ? bout : bin;
        const int    mulbase    = phase ? 0 : 128;   // gin*fd , gout*fs
        const int*   scat       = phase ? didx : sidx;
        const int    k8         = phase ? 288 : 256; // LDS col of 9th K-tile

        v8f av[2][8];
#pragma unroll
        for (int i = 0; i < 2; ++i)
#pragma unroll
            for (int j = 0; j < 8; ++j)
#pragma unroll
                for (int l = 0; l < 8; ++l) av[i][j][l] = 0.f;

        for (int kt = 0; kt < 9; ++kt) {
            int klds = (kt < 8) ? kt * 32 : k8;
            frag32 a0, a1;                           // ds_load_b128 x4
            a0.q[0] = *(const uint4*)&feat[mloc][klds + hi * 8];
            a0.q[1] = *(const uint4*)&feat[mloc][klds + 16 + hi * 8];
            a1.q[0] = *(const uint4*)&feat[16 + mloc][klds + hi * 8];
            a1.q[1] = *(const uint4*)&feat[16 + mloc][klds + 16 + hi * 8];
            frag32 b[8];                             // 16x b128, loaded up front
            const unsigned short* bp = B + (((size_t)(kt * 8) * 32 + lane) << 4);
#pragma unroll
            for (int nt = 0; nt < 8; ++nt) {
                b[nt].q[0] = *(const uint4*)(bp);
                b[nt].q[1] = *(const uint4*)(bp + 8);
                bp += 512;
            }
#pragma unroll
            for (int nt = 0; nt < 8; ++nt) {
                av[0][nt] = wmma_bf16(a0.v, b[nt].v, av[0][nt]);
                av[1][nt] = wmma_bf16(a1.v, b[nt].v, av[1][nt]);
            }
        }
#pragma unroll
        for (int mt = 0; mt < 2; ++mt)
#pragma unroll
            for (int nt = 0; nt < 8; ++nt) {
                int ch = nt * 16 + mloc;
                float bv = bias[ch];
#pragma unroll
                for (int r = 0; r < 8; ++r) {
                    int em = mt * 16 + r + hi * 8;
                    float g = sigm(av[mt][nt][r] + bv);
                    float mul = bfu2f(feat[em][mulbase + ch]);
                    atomicAdd(&acc[(size_t)scat[em] * 128 + ch], g * mul);
                }
            }
    }
}

// ---------------- LSTM recurrence (one block per batch row) ----------------
__global__ __launch_bounds__(64) void k_lstm_rec(
    const float* __restrict__ pre,   // B*L x 256 (already + bih + bhh)
    const float* __restrict__ Whh,   // 256 x 64
    float* __restrict__ hout,        // B*L x 64
    int reverse)
{
    int b = blockIdx.x, j = threadIdx.x;
    __shared__ float h[HID];
    float c = 0.f;
    h[j] = 0.f;
    __syncthreads();
    for (int s = 0; s < SEQL; ++s) {
        int t = reverse ? (SEQL - 1 - s) : s;
        const float* p = pre + (size_t)(b * SEQL + t) * 256;
        float gi = p[j], gf = p[64 + j], gg = p[128 + j], go = p[192 + j];
        for (int k = 0; k < HID; ++k) {
            float hk = h[k];
            gi += Whh[j * HID + k] * hk;
            gf += Whh[(64 + j) * HID + k] * hk;
            gg += Whh[(128 + j) * HID + k] * hk;
            go += Whh[(192 + j) * HID + k] * hk;
        }
        c = sigm(gf) * c + sigm(gi) * tanhf(gg);
        float hn = sigm(go) * tanhf(c);
        __syncthreads();
        h[j] = hn;
        __syncthreads();
        hout[(size_t)(b * SEQL + t) * HID + j] = hn;
    }
}

// ---------------- layernorms ----------------
__global__ __launch_bounds__(128) void k_seq_ln(
    const float* __restrict__ s, const float* __restrict__ hf, const float* __restrict__ hb,
    const float* __restrict__ g, const float* __restrict__ bb, float* __restrict__ out)
{
    int r = blockIdx.x, d = threadIdx.x;
    float v = s[(size_t)r * 128 + d] + (d < 64 ? hf[(size_t)r * 64 + d] : hb[(size_t)r * 64 + d - 64]);
    __shared__ float red[128];
    red[d] = v; __syncthreads();
    for (int st = 64; st > 0; st >>= 1) { if (d < st) red[d] += red[d + st]; __syncthreads(); }
    float mean = red[0] * (1.f / 128.f);
    __syncthreads();
    float dv = v - mean;
    red[d] = dv * dv; __syncthreads();
    for (int st = 64; st > 0; st >>= 1) { if (d < st) red[d] += red[d + st]; __syncthreads(); }
    float var = red[0] * (1.f / 128.f);
    out[(size_t)r * 128 + d] = dv * rsqrtf(var + LN_EPS) * g[d] + bb[d];
}

__global__ __launch_bounds__(128) void k_ln(
    const float* __restrict__ x, const float* __restrict__ g,
    const float* __restrict__ bb, float* __restrict__ out)
{
    int r = blockIdx.x, d = threadIdx.x;
    float v = x[(size_t)r * 128 + d];
    __shared__ float red[128];
    red[d] = v; __syncthreads();
    for (int st = 64; st > 0; st >>= 1) { if (d < st) red[d] += red[d + st]; __syncthreads(); }
    float mean = red[0] * (1.f / 128.f);
    __syncthreads();
    float dv = v - mean;
    red[d] = dv * dv; __syncthreads();
    for (int st = 64; st > 0; st >>= 1) { if (d < st) red[d] += red[d + st]; __syncthreads(); }
    float var = red[0] * (1.f / 128.f);
    out[(size_t)r * 128 + d] = dv * rsqrtf(var + LN_EPS) * g[d] + bb[d];
}

// ---------------- elementwise / prep ----------------
__global__ void k_add3(const float* a, const float* b, const float* c, float* o, int n) {
    int i = blockIdx.x * blockDim.x + threadIdx.x;
    if (i < n) o[i] = a[i] + b[i] + c[i];
}
__global__ void k_add2(const float* a, const float* b, float* o, int n) {
    int i = blockIdx.x * blockDim.x + threadIdx.x;
    if (i < n) o[i] = a[i] + b[i];
}
// fp32 (ldsrc) -> bf16 (lddst) with zero padding beyond Keff
__global__ void k_cvt_pad(const float* __restrict__ src, int ldsrc,
                          unsigned short* __restrict__ dst, int lddst,
                          int rows, int Keff) {
    int i = blockIdx.x * blockDim.x + threadIdx.x;
    if (i < rows * lddst) {
        int r = i / lddst, c = i % lddst;
        dst[i] = (c < Keff) ? f2bfu(src[(size_t)r * ldsrc + c]) : (unsigned short)0;
    }
}
// Weight -> fragment-order bf16: [kt][nt][lane][16], k = kt*32 + (e>>3)*16 + (lane>>4)*8 + (e&7),
// n = nt*16 + (lane&15). W is (Nout x Keff) row-major.
__global__ void k_prepB(const float* __restrict__ W, unsigned short* __restrict__ B,
                        int Keff, int KT, int NT) {
    int i = blockIdx.x * blockDim.x + threadIdx.x;
    if (i < KT * NT * 512) {
        int e    = i & 15;
        int lane = (i >> 4) & 31;
        int blk  = i >> 9;
        int nt   = blk % NT, kt = blk / NT;
        int n = nt * 16 + (lane & 15);
        int k = kt * 32 + (e >> 3) * 16 + (lane >> 4) * 8 + (e & 7);
        B[i] = (k < Keff) ? f2bfu(W[(size_t)n * Keff + k]) : (unsigned short)0;
    }
}
__global__ void k_bias2(const float* a, const float* b, float* o, int n) {
    int i = threadIdx.x;
    if (i < n) o[i] = a[i] + b[i];
}
// structA (bf16, 16384 x 160): [esm | x[:,0] | zeros]
__global__ void k_structA(const float* __restrict__ esm, const float* __restrict__ x,
                          unsigned short* __restrict__ A) {
    int i = blockIdx.x * blockDim.x + threadIdx.x;
    if (i < NNODES * 160) {
        int r = i / 160, c = i % 160;
        float v = (c < 128) ? esm[(size_t)r * 128 + c]
                : (c == 128) ? x[(size_t)r * 2561] : 0.f;
        A[i] = f2bfu(v);
    }
}
__global__ void k_starts(const int* batch, int* starts) {
    int b = threadIdx.x;
    if (b < BATCHN) {
        int lo = 0, hi = NNODES;
        while (lo < hi) { int mid = (lo + hi) >> 1; if (batch[mid] < b) lo = mid + 1; else hi = mid; }
        starts[b] = lo;
    }
}
__global__ __launch_bounds__(128) void k_final(
    const float* __restrict__ sy, const float* __restrict__ ty, const int* __restrict__ starts,
    const float* __restrict__ rm, const float* __restrict__ rv,
    const float* __restrict__ g, const float* __restrict__ bb, float* __restrict__ out)
{
    int b = blockIdx.x, d = threadIdx.x;
    float feat = sy[(size_t)(b * SEQL + 10) * 128 + d] + ty[(size_t)starts[b] * 128 + d];
    out[(size_t)b * 128 + d] = (feat - rm[d]) * rsqrtf(rv[d] + LN_EPS) * g[d] + bb[d];
}

// ---------------- host orchestration ----------------
extern "C" void kernel_launch(void* const* d_in, const int* in_sizes, int n_in,
                              void* d_out, int out_size, void* d_ws, size_t ws_size,
                              hipStream_t stream) {
    (void)in_sizes; (void)n_in; (void)out_size; (void)ws_size;
    const float* x          = (const float*)d_in[0];
    const int*   edge_index = (const int*)  d_in[1];
    const float* edge_attr  = (const float*)d_in[2];
    const int*   batch      = (const int*)  d_in[3];
    const float* hotslice   = (const float*)d_in[4];
    const float* n_sy       = (const float*)d_in[5];
    const float* n_sz       = (const float*)d_in[6];
    const float* n_ty       = (const float*)d_in[7];
    const float* n_tz       = (const float*)d_in[8];
    const float* W_esm      = (const float*)d_in[9];
    const float* b_esm      = (const float*)d_in[10];
    const float* W_node     = (const float*)d_in[11];
    const float* b_node     = (const float*)d_in[12];
    const float* W_seq      = (const float*)d_in[13];
    const float* b_seq      = (const float*)d_in[14];
    const float* ln_seq_g   = (const float*)d_in[15];
    const float* ln_seq_b   = (const float*)d_in[16];
    const float* ln_st_g    = (const float*)d_in[17];
    const float* ln_st_b    = (const float*)d_in[18];
    const float* W_in       = (const float*)d_in[19];
    const float* b_in       = (const float*)d_in[20];
    const float* W_out      = (const float*)d_in[21];
    const float* b_out      = (const float*)d_in[22];
    const float* Wih_f      = (const float*)d_in[23];
    const float* Whh_f      = (const float*)d_in[24];
    const float* bih_f      = (const float*)d_in[25];
    const float* bhh_f      = (const float*)d_in[26];
    const float* Wih_b      = (const float*)d_in[27];
    const float* Whh_b      = (const float*)d_in[28];
    const float* bih_b      = (const float*)d_in[29];
    const float* bhh_b      = (const float*)d_in[30];
    const float* bn_g       = (const float*)d_in[31];
    const float* bn_b       = (const float*)d_in[32];
    const float* bn_rm      = (const float*)d_in[33];
    const float* bn_rv      = (const float*)d_in[34];
    float* out = (float*)d_out;

    // ---- carve workspace ----
    char* p = (char*)d_ws;
    auto alloc = [&](size_t bytes) -> void* {
        void* r = (void*)p;
        p += (bytes + 255) & ~(size_t)255;
        return r;
    };
    float* seq0    = (float*)alloc(SEQSZ * 4);
    float* sy      = (float*)alloc(SEQSZ * 4);
    float* sz      = (float*)alloc(SEQSZ * 4);
    float* seq_in  = (float*)alloc(SEQSZ * 4);
    float* struct0 = (float*)alloc((size_t)NODESZ * 4);
    float* ty      = (float*)alloc((size_t)NODESZ * 4);
    float* tz      = (float*)alloc((size_t)NODESZ * 4);
    float* st_in   = (float*)alloc((size_t)NODESZ * 4);
    float* st_acc  = (float*)alloc((size_t)NODESZ * 4);
    float* esm     = (float*)alloc((size_t)NODESZ * 4);
    float* pre_f   = (float*)alloc(GATESZ * 4);
    float* pre_b   = (float*)alloc(GATESZ * 4);
    float* hf      = (float*)alloc(HBUFSZ * 4);
    float* hb      = (float*)alloc(HBUFSZ * 4);
    float* bias_f  = (float*)alloc(256 * 4);
    float* bias_b  = (float*)alloc(256 * 4);
    int*   starts  = (int*)  alloc(64 * 4);
    // bf16 A operands (zero-padded, lda % 8 == 0)
    unsigned short* xbf      = (unsigned short*)alloc((size_t)NNODES * 2560 * 2);
    unsigned short* structAb = (unsigned short*)alloc((size_t)NNODES * 160 * 2);
    unsigned short* hotbf    = (unsigned short*)alloc((size_t)SEQROWS * 32 * 2);
    unsigned short* seqbf    = (unsigned short*)alloc((size_t)SEQSZ * 2);
    unsigned short* nodesb   = (unsigned short*)alloc((size_t)NODESZ * 2);
    // bf16 fragment-order weights
    unsigned short* Besm  = (unsigned short*)alloc((size_t)80 * 8 * 512 * 2);
    unsigned short* Bnode = (unsigned short*)alloc((size_t)5 * 8 * 512 * 2);
    unsigned short* Bseq  = (unsigned short*)alloc((size_t)1 * 8 * 512 * 2);
    unsigned short* Bih_f = (unsigned short*)alloc((size_t)4 * 16 * 512 * 2);
    unsigned short* Bih_b = (unsigned short*)alloc((size_t)4 * 16 * 512 * 2);
    unsigned short* Bin   = (unsigned short*)alloc((size_t)9 * 8 * 512 * 2);
    unsigned short* Bout  = (unsigned short*)alloc((size_t)9 * 8 * 512 * 2);

    // ---- weight prep ----
    auto gprep = [&](const float* W, unsigned short* B, int Keff, int KT, int NT) {
        int n = KT * NT * 512;
        k_prepB<<<(n + 255) / 256, 256, 0, stream>>>(W, B, Keff, KT, NT);
    };
    gprep(W_esm,  Besm,  2560, 80, 8);
    gprep(W_node, Bnode, 129,  5,  8);
    gprep(W_seq,  Bseq,  28,   1,  8);
    gprep(Wih_f,  Bih_f, 128,  4,  16);
    gprep(Wih_b,  Bih_b, 128,  4,  16);
    gprep(W_in,   Bin,   259,  9,  8);
    gprep(W_out,  Bout,  259,  9,  8);
    k_bias2<<<1, 256, 0, stream>>>(bih_f, bhh_f, bias_f, 256);
    k_bias2<<<1, 256, 0, stream>>>(bih_b, bhh_b, bias_b, 256);

    // ---- A-operand conversions (one-time) ----
    {
        size_t n = (size_t)NNODES * 2560;
        k_cvt_pad<<<(n + 255) / 256, 256, 0, stream>>>(x + 1, 2561, xbf, 2560, NNODES, 2560);
        int m = SEQROWS * 32;
        k_cvt_pad<<<(m + 255) / 256, 256, 0, stream>>>(hotslice, 28, hotbf, 32, SEQROWS, 28);
    }

    // ---- one-time GEMMs ----
    // esm = x[:,1:] @ W_esm.T + b_esm
    k_gemm<<<dim3(NNODES / 32, 1), 32, 0, stream>>>(xbf, 2560, 2560, Besm, 8, b_esm, esm, 128);
    // struct0 = [esm | x[:,:1]] @ W_node.T + b_node
    k_structA<<<(NNODES * 160 + 255) / 256, 256, 0, stream>>>(esm, x, structAb);
    k_gemm<<<dim3(NNODES / 32, 1), 32, 0, stream>>>(structAb, 160, 160, Bnode, 8, b_node, struct0, 128);
    // seq0 = hotslice @ W_seq.T + b_seq
    k_gemm<<<dim3(SEQROWS / 32, 1), 32, 0, stream>>>(hotbf, 32, 32, Bseq, 8, b_seq, seq0, 128);

    // ---- init carry from noise ----
    hipMemcpyAsync(sy, n_sy, SEQSZ * 4, hipMemcpyDeviceToDevice, stream);
    hipMemcpyAsync(sz, n_sz, SEQSZ * 4, hipMemcpyDeviceToDevice, stream);
    hipMemcpyAsync(ty, n_ty, (size_t)NODESZ * 4, hipMemcpyDeviceToDevice, stream);
    hipMemcpyAsync(tz, n_tz, (size_t)NODESZ * 4, hipMemcpyDeviceToDevice, stream);

    // ---- one ss(seq_in, st_in) -> (seq_out, st_out) ----
    auto ss = [&](float* seq_i, float* st_i, float* seq_o, float* st_o) {
        // BiLSTM input projections (WMMA, bf16 A)
        k_cvt_pad<<<(SEQSZ + 255) / 256, 256, 0, stream>>>(seq_i, 128, seqbf, 128, SEQROWS, 128);
        k_gemm<<<dim3(SEQROWS / 32, 2), 32, 0, stream>>>(seqbf, 128, 128, Bih_f, 16, bias_f, pre_f, 256);
        k_gemm<<<dim3(SEQROWS / 32, 2), 32, 0, stream>>>(seqbf, 128, 128, Bih_b, 16, bias_b, pre_b, 256);
        k_lstm_rec<<<BATCHN, HID, 0, stream>>>(pre_f, Whh_f, hf, 0);
        k_lstm_rec<<<BATCHN, HID, 0, stream>>>(pre_b, Whh_b, hb, 1);
        k_seq_ln<<<SEQROWS, 128, 0, stream>>>(seq_i, hf, hb, ln_seq_g, ln_seq_b, seq_o);
        // edge gating (fused WMMA gather/scatter)
        k_cvt_pad<<<(NODESZ + 255) / 256, 256, 0, stream>>>(st_i, 128, nodesb, 128, NNODES, 128);
        hipMemcpyAsync(st_acc, st_i, (size_t)NODESZ * 4, hipMemcpyDeviceToDevice, stream);
        k_edge<<<EHALF / 32, 32, 0, stream>>>(nodesb, edge_index, edge_attr,
                                              Bin, Bout, b_in, b_out, st_acc);
        k_ln<<<NNODES, 128, 0, stream>>>(st_acc, ln_st_g, ln_st_b, st_o);
    };

    // ---- main iteration: OUTSIDE=8 outer, each with INSIDE=4 inner + 1 ----
    for (int o = 0; o < 8; ++o) {
        for (int in = 0; in < 4; ++in) {
            k_add3<<<(SEQSZ + 255) / 256, 256, 0, stream>>>(seq0, sy, sz, seq_in, SEQSZ);
            k_add3<<<(NODESZ + 255) / 256, 256, 0, stream>>>(struct0, ty, tz, st_in, NODESZ);
            ss(seq_in, st_in, sz, tz);
        }
        k_add2<<<(SEQSZ + 255) / 256, 256, 0, stream>>>(sy, sz, seq_in, SEQSZ);
        k_add2<<<(NODESZ + 255) / 256, 256, 0, stream>>>(ty, tz, st_in, NODESZ);
        ss(seq_in, st_in, sy, ty);
    }

    // ---- readout ----
    k_starts<<<1, 64, 0, stream>>>(batch, starts);
    k_final<<<BATCHN, 128, 0, stream>>>(sy, ty, starts, bn_rm, bn_rv, bn_g, bn_b, out);
}
